// ChebConvReal_37056977830139
// MI455X (gfx1250) — matrix operand order repaired and checked by reading, not compiled, not run on hardware
//
#include <hip/hip_runtime.h>
#include <hip/hip_bf16.h>

#define BB 8
#define NN 1024
#define CC 128
#define OO 128
#define KK 4   // K+1 Chebyshev orders

typedef __attribute__((ext_vector_type(16))) __bf16 v16bf;
typedef __attribute__((ext_vector_type(8)))  float  v8f;

union BFrag16 { v16bf v; unsigned int u[8]; uint4 q[2]; };
union CFrag   { v8f   v; float f[8]; };

__device__ __forceinline__ unsigned short f2bf(float f) {
    union { float f; unsigned int u; } a; a.f = f;
    unsigned int r = a.u + 0x7FFFu + ((a.u >> 16) & 1u);
    return (unsigned short)(r >> 16);
}
__device__ __forceinline__ unsigned int pack2bf(float lo, float hi) {
    return (unsigned int)f2bf(lo) | ((unsigned int)f2bf(hi) << 16);
}

// ---------------------------------------------------------------------------
// Kernel 1: bf16-convert x and weight, pre-swizzled into WMMA B-fragment
// layout so k_main lane loads are two global_load_b128 per fragment.
// B 32x16 bf16 layout (contiguous k = 16h + s order used by this code, not the ISA table order): lane = (row>=16)*16 + col ; VGPR v holds
// packed rows {2v,2v+1} of the lane's half. Fragment stored lane-major:
// ushort offset = lane*16 + v*2 + (row&1).
// ---------------------------------------------------------------------------
__global__ void k_convert(const float* __restrict__ x, const float* __restrict__ w,
                          unsigned short* __restrict__ xf, unsigned short* __restrict__ wf) {
    int i = blockIdx.x * blockDim.x + threadIdx.x;
    if (i < BB * NN * CC) {
        int c  = i & (CC - 1);
        int m  = (i / CC) & (NN - 1);
        int b  = i / (CC * NN);
        int mt = m >> 5, row = m & 31;
        int ch = c >> 4, col = c & 15;
        int lanef = ((row >> 4) << 4) | col;
        size_t d = (((size_t)(b * (NN >> 5) + mt) * (CC >> 4) + ch) << 9)
                 + lanef * 16 + (((row & 15) >> 1) << 1) + (row & 1);
        xf[d] = f2bf(x[i]);
    }
    if (i < KK * CC * OO) {
        int o  = i & (OO - 1);
        int c  = (i / OO) & (CC - 1);
        int k  = i / (OO * CC);
        int cs = c >> 5, row = c & 31;
        int ch = o >> 4, col = o & 15;
        int lanef = ((row >> 4) << 4) | col;
        size_t d = (((size_t)(k * (CC >> 5) + cs) * (OO >> 4) + ch) << 9)
                 + lanef * 16 + (((row & 15) >> 1) << 1) + (row & 1);
        wf[d] = f2bf(w[i]);
    }
}

// ---------------------------------------------------------------------------
// Kernel 2: si[b,n] = x[b,n,:]·w_i ; tj[b,n] = x[b,n,:]·w_j  (wave per row)
// ---------------------------------------------------------------------------
__global__ void k_scores(const float* __restrict__ x, const float* __restrict__ attw,
                         float* __restrict__ si, float* __restrict__ tj) {
    int wave = (blockIdx.x * blockDim.x + threadIdx.x) >> 5;
    int lane = threadIdx.x & 31;
    if (wave >= BB * NN) return;
    const float4 xr = *(const float4*)&x[(size_t)wave * CC + lane * 4];
    const float4 wi = *(const float4*)&attw[lane * 4];
    const float4 wj = *(const float4*)&attw[CC + lane * 4];
    float a = xr.x * wi.x + xr.y * wi.y + xr.z * wi.z + xr.w * wi.w;
    float b = xr.x * wj.x + xr.y * wj.y + xr.z * wj.z + xr.w * wj.w;
    #pragma unroll
    for (int off = 16; off > 0; off >>= 1) {
        a += __shfl_down(a, off, 32);
        b += __shfl_down(b, off, 32);
    }
    if (lane == 0) { si[wave] = a; tj[wave] = b; }
}

// ---------------------------------------------------------------------------
// Kernel 3: per (b,n) softmax statistics: row max and 1/sum(exp)
// ---------------------------------------------------------------------------
__global__ void k_stats(const float* __restrict__ si, const float* __restrict__ tj,
                        const float* __restrict__ attn_b,
                        float* __restrict__ rowmax, float* __restrict__ rowinv) {
    __shared__ float red[256];
    const int row = blockIdx.x;          // b*NN + n
    const int b   = row >> 10;
    const float ab = attn_b[0];
    const float sv = si[row];
    float mx = -3.4e38f;
    for (int m = threadIdx.x; m < NN; m += 256) {
        float s = sv + tj[b * NN + m] + ab;
        s = s >= 0.f ? s : 0.2f * s;
        mx = fmaxf(mx, s);
    }
    red[threadIdx.x] = mx; __syncthreads();
    for (int o = 128; o > 0; o >>= 1) {
        if (threadIdx.x < o) red[threadIdx.x] = fmaxf(red[threadIdx.x], red[threadIdx.x + o]);
        __syncthreads();
    }
    mx = red[0]; __syncthreads();
    float sum = 0.f;
    for (int m = threadIdx.x; m < NN; m += 256) {
        float s = sv + tj[b * NN + m] + ab;
        s = s >= 0.f ? s : 0.2f * s;
        sum += __expf(s - mx);
    }
    red[threadIdx.x] = sum; __syncthreads();
    for (int o = 128; o > 0; o >>= 1) {
        if (threadIdx.x < o) red[threadIdx.x] += red[threadIdx.x + o];
        __syncthreads();
    }
    if (threadIdx.x == 0) { rowmax[row] = mx; rowinv[row] = 1.f / red[0]; }
}

// ---------------------------------------------------------------------------
// Kernel 4: main fused kernel. One workgroup (8 waves) per 16-row n-tile.
// Double-buffered A tiles in LDS -> single barrier per 32-wide m-step.
// ---------------------------------------------------------------------------
__global__ void __launch_bounds__(256, 2)
k_main(const float* __restrict__ lap, const unsigned short* __restrict__ xf,
       const unsigned short* __restrict__ wf, const float* __restrict__ si,
       const float* __restrict__ tj, const float* __restrict__ rowmax,
       const float* __restrict__ rowinv, const float* __restrict__ attn_b,
       const float* __restrict__ bias, float* __restrict__ out)
{
    __shared__ unsigned int   sA[2][KK][16][16];   // double-buffered bf16 A tiles (8 KB)
    __shared__ unsigned short sLx[KK][16][CC];     // lx staged as bf16 (16 KB)

    const int tid  = threadIdx.x;
    const int lane = tid & 31;
    const int wave = tid >> 5;
    const int b    = blockIdx.x >> 6;
    const int n0   = (blockIdx.x & 63) << 4;
    const float ab = attn_b[0];

    // this thread's laplacian quad: row rr, columns mm..mm+3, k in {kb, kb+1}
    const int q  = tid & 127;
    const int rr = q >> 3;
    const int mm = (q & 7) << 2;
    const int kb = (tid >> 7) << 1;

    // per-thread softmax state for row rr (broadcast loads, L2-resident)
    const float svr = si[b * NN + n0 + rr];
    const float mxr = rowmax[b * NN + n0 + rr];
    const float ivr = rowinv[b * NN + n0 + rr];

    // WMMA fragment lane geometry (ISA 7.12.2, wave32)
    const int am  = lane & 15;            // A: M row
    const int ahi = lane >> 4;            // A: K half select

    v8f accs[KK] = {};

    const unsigned short* xfb = xf + (((size_t)b * (NN >> 5) * (CC >> 4) + wave) << 9);
    const float* lapr = lap + (((size_t)(b * KK + kb)) * NN + (n0 + rr)) * NN + mm;

    for (int it = 0; it < (NN >> 5); ++it) {
        const int m0 = it << 5;
        const int p  = it & 1;

        // ---- attention quad (f32, recomputed in-thread: no LDS round trip) ----
        const float4 tv = *(const float4*)&tj[b * NN + m0 + mm];
        float s0 = svr + tv.x + ab; s0 = s0 >= 0.f ? s0 : 0.2f * s0;
        float s1 = svr + tv.y + ab; s1 = s1 >= 0.f ? s1 : 0.2f * s1;
        float s2 = svr + tv.z + ab; s2 = s2 >= 0.f ? s2 : 0.2f * s2;
        float s3 = svr + tv.w + ab; s3 = s3 >= 0.f ? s3 : 0.2f * s3;
        const float a0 = __expf(s0 - mxr) * ivr;
        const float a1 = __expf(s1 - mxr) * ivr;
        const float a2 = __expf(s2 - mxr) * ivr;
        const float a3 = __expf(s3 - mxr) * ivr;

        // ---- lap * attn -> bf16 A tiles (two k's per thread) ----
        #pragma unroll
        for (int kk = 0; kk < 2; ++kk) {
            const float4 lv = *(const float4*)(lapr + (size_t)kk * NN * NN + m0);
            sA[p][kb + kk][rr][mm >> 1]       = pack2bf(lv.x * a0, lv.y * a1);
            sA[p][kb + kk][rr][(mm >> 1) + 1] = pack2bf(lv.z * a2, lv.w * a3);
        }
        // prefetch next laplacian tile (gfx1250 global_prefetch_b8)
        if (m0 + 32 < NN) {
            __builtin_prefetch(lapr + m0 + 32, 0, 1);
            __builtin_prefetch(lapr + (size_t)NN * NN + m0 + 32, 0, 1);
        }

        // ---- B fragment: pre-swizzled xf, two global_load_b128 ----
        BFrag16 bfx;
        {
            const unsigned short* p8 = xfb + ((size_t)it * (CC >> 4) << 9) + lane * 16;
            bfx.q[0] = *(const uint4*)(p8);
            bfx.q[1] = *(const uint4*)(p8 + 8);
        }

        __syncthreads();

        // ---- A fragments from LDS + 4 WMMAs ----
        #pragma unroll
        for (int k = 0; k < KK; ++k) {
            BFrag16 af;
            #pragma unroll
            for (int v = 0; v < 8; ++v) {
                int ki = v + (ahi << 3);   // packed K-pair index
                af.u[v] = sA[p][k][am][ki];
            }
            accs[k] = __builtin_amdgcn_wmma_f32_16x16x32_bf16(false, af.v, false, bfx.v,
                                                              (short)0, accs[k], false, false);
        }
    }

    // ---- stage lx as bf16 in LDS ----
    const int c0 = wave << 4;
    #pragma unroll
    for (int k = 0; k < KK; ++k) {
        CFrag cf; cf.v = accs[k];
        #pragma unroll
        for (int j = 0; j < 8; ++j) {
            int r = j + ((lane >> 4) << 3);
            sLx[k][r][c0 + (lane & 15)] = f2bf(cf.f[j]);
        }
    }
    __syncthreads();

    // ---- stage 2: out[16, o0..o0+15] = sum_k lx_k @ W_k ----
    const int o0 = wave << 4;
    v8f oacc = {};
    #pragma unroll
    for (int k = 0; k < KK; ++k) {
        #pragma unroll
        for (int cs = 0; cs < (CC >> 5); ++cs) {
            BFrag16 af, bfw;
            #pragma unroll
            for (int v = 0; v < 8; ++v) {
                int ki = (cs << 5) + 2 * (v + (ahi << 3));
                af.u[v] = *(const unsigned int*)&sLx[k][am][ki];
            }
            {
                const unsigned short* p8 = wf + (((size_t)(k * (CC >> 5) + cs) * (OO >> 4) + wave) << 9)
                                         + lane * 16;
                bfw.q[0] = *(const uint4*)(p8);
                bfw.q[1] = *(const uint4*)(p8 + 8);
            }
            oacc = __builtin_amdgcn_wmma_f32_16x16x32_bf16(false, af.v, false, bfw.v,
                                                           (short)0, oacc, false, false);
        }
    }

    // ---- epilogue: add bias, store f32 ----
    {
        float bv = bias[o0 + (lane & 15)];
        CFrag cf; cf.v = oacc;
        #pragma unroll
        for (int j = 0; j < 8; ++j) {
            int r = j + ((lane >> 4) << 3);
            out[((size_t)b * NN + n0 + r) * OO + o0 + (lane & 15)] = cf.f[j] + bv;
        }
    }
}

// ---------------------------------------------------------------------------
extern "C" void kernel_launch(void* const* d_in, const int* in_sizes, int n_in,
                              void* d_out, int out_size, void* d_ws, size_t ws_size,
                              hipStream_t stream) {
    const float* x    = (const float*)d_in[0];
    const float* lap  = (const float*)d_in[1];
    const float* wgt  = (const float*)d_in[2];
    const float* attw = (const float*)d_in[3];
    const float* attb = (const float*)d_in[4];
    const float* bias = (const float*)d_in[5];
    float* out = (float*)d_out;

    char* ws = (char*)d_ws;
    float*          si   = (float*)(ws);                              // 32 KB
    float*          tj   = (float*)(ws + 32768);                      // 32 KB
    float*          rmax = (float*)(ws + 65536);                      // 32 KB
    float*          rinv = (float*)(ws + 98304);                      // 32 KB
    unsigned short* xf   = (unsigned short*)(ws + 131072);            // 2 MB, swizzled
    unsigned short* wf   = (unsigned short*)(ws + 131072 + 2097152);  // 128 KB, swizzled

    k_convert<<<(BB * NN * CC + 255) / 256, 256, 0, stream>>>(x, wgt, xf, wf);
    k_scores<<<(BB * NN * 32 + 255) / 256, 256, 0, stream>>>(x, attw, si, tj);
    k_stats<<<BB * NN, 256, 0, stream>>>(si, tj, attb, rmax, rinv);
    k_main<<<BB * (NN / 16), 256, 0, stream>>>(lap, xf, wf, si, tj, rmax, rinv, attb, bias, out);
}
